// MultiHeadAttention_32023276159092
// MI455X (gfx1250) — matrix operand order repaired and checked by reading, not compiled, not run on hardware
//
#include <hip/hip_runtime.h>
#include <hip/hip_bf16.h>

// ---------------- problem constants ----------------
constexpr int B_ = 4;
constexpr int T_ = 1024;
constexpr int D_ = 2048;
constexpr int H_ = 16;
constexpr int E_ = 128;  // head dim

typedef __attribute__((ext_vector_type(16))) _Float16 v16h;
typedef __attribute__((ext_vector_type(8)))  _Float16 v8h;
typedef __attribute__((ext_vector_type(8)))  float    v8f;
typedef __attribute__((ext_vector_type(4)))  int      v4i;

#define NEG_INF (-__builtin_inff())

static __device__ __forceinline__ v16h combine8(v8h lo, v8h hi) {
  return __builtin_shufflevector(lo, hi, 0,1,2,3,4,5,6,7,8,9,10,11,12,13,14,15);
}

static __device__ __forceinline__ v8f wmma_f16(v16h a, v16h b, v8f c) {
  // D = A(16xK=32) * B(32x16) + C, f32 accumulate
  return __builtin_amdgcn_wmma_f32_16x16x32_f16(
      /*neg_a=*/false, a, /*neg_b=*/false, b,
      /*c_mod=*/(short)0, c, /*reuse_a=*/false, /*reuse_b=*/false);
}

// Async 16-byte copy global -> LDS (ASYNCcnt-tracked).  Generic pointers to
// LDS carry the LDS byte offset in their low 32 bits.
static __device__ __forceinline__ void async_copy_b128(const _Float16* g,
                                                       _Float16* l) {
  unsigned long long ga = (unsigned long long)(uintptr_t)g;
  unsigned int la = (unsigned int)(uintptr_t)l;
  asm volatile("global_load_async_to_lds_b128 %0, %1, off"
               :: "v"(la), "v"(ga) : "memory");
}

static __device__ __forceinline__ void wait_async_le1() {
  asm volatile("s_wait_asynccnt 0x1" ::: "memory");
}
static __device__ __forceinline__ void wait_async_le0() {
  asm volatile("s_wait_asynccnt 0x0" ::: "memory");
}

// ---------------- f32 -> f16 convert ----------------
__global__ void cvt_f16_kernel(const float* __restrict__ src,
                               _Float16* __restrict__ dst, int n) {
  int i = blockIdx.x * blockDim.x + threadIdx.x;
  int stride = gridDim.x * blockDim.x;
  for (; i < n; i += stride) dst[i] = (_Float16)src[i];
}

// ---------------- per-head QKV projection GEMM ----------------
// Block = 8 waves computing a 128(M=t) x 64(N=e) tile; B panel (64e x 32k)
// staged into LDS with async loads, double-buffered, shared by all waves.
// out[t,e] = sum_d X[b,t,d] * W[h,e,d] + bias[h,e]
// transposed==0 : store [b][h][t][e] (Q,K);  ==1 : store [b][h][e][t] (V)
__global__ __launch_bounds__(256)
void qkv_gemm_kernel(const _Float16* __restrict__ X,
                     const _Float16* __restrict__ W,
                     const float* __restrict__ bias,
                     _Float16* __restrict__ out,
                     int transposed) {
  __shared__ alignas(64) _Float16 bt[2][64 * 32];

  const int tid = threadIdx.x;
  const int wv = tid >> 5, lane = tid & 31;
  const int ls = lane & 15, hi = lane >> 4;

  const int nb = blockIdx.x & 1;          // E/64 = 2
  const int mb = (blockIdx.x >> 1) & 7;   // T/128 = 8
  const int h  = (blockIdx.x >> 4) & 15;
  const int b  = blockIdx.x >> 8;

  const int n0 = nb * 64;
  const int m0 = mb * 128 + wv * 16;

  // this thread's slice of the B panel copy: row e = n0 + tid/4, 16B segment
  const int crow = tid >> 2, cseg = tid & 3;
  const _Float16* gsrc = W + ((size_t)h * E_ + n0 + crow) * D_ + cseg * 8;
  _Float16* l0 = &bt[0][crow * 32 + cseg * 8];
  _Float16* l1 = &bt[1][crow * 32 + cseg * 8];

  const _Float16* arow = X + ((size_t)(b * T_) + m0 + ls) * D_;

  v8f acc[4] = {};
  constexpr int STEPS = D_ / 32;  // 64

  async_copy_b128(gsrc, l0);
  for (int s = 0; s < STEPS; ++s) {
    const int cur = s & 1;
    const bool more = (s + 1) < STEPS;
    if (more) async_copy_b128(gsrc + (size_t)(s + 1) * 32, cur ? l0 : l1);
    if (more) wait_async_le1(); else wait_async_le0();
    __syncthreads();  // current panel visible to all waves

    const int k0 = s * 32;
    v8h alo = *(const v8h*)(arow + k0 + hi * 16);
    v8h ahi = *(const v8h*)(arow + k0 + hi * 16 + 8);
    v16h a  = combine8(alo, ahi);
#pragma unroll
    for (int nt = 0; nt < 4; ++nt) {
      v16h bb = *(const v16h*)&bt[cur][(nt * 16 + ls) * 32 + hi * 16];
      acc[nt] = wmma_f16(a, bb, acc[nt]);
    }
    __syncthreads();  // all reads of bt[cur] done before step s+2 overwrites
  }

#pragma unroll
  for (int nt = 0; nt < 4; ++nt) {
    const int e = n0 + nt * 16 + ls;
    const float bval = bias[h * E_ + e];
    if (!transposed) {
      _Float16* op = out + (((size_t)(b * H_ + h)) * T_ + m0 + hi * 8) * E_ + e;
#pragma unroll
      for (int j = 0; j < 8; ++j) op[(size_t)j * E_] = (_Float16)(acc[nt][j] + bval);
    } else {
      v8h o;
#pragma unroll
      for (int j = 0; j < 8; ++j) o[j] = (_Float16)(acc[nt][j] + bval);
      *(v8h*)(out + (((size_t)(b * H_ + h)) * E_ + e) * T_ + m0 + hi * 8) = o;
    }
  }
}

// ---------------- flash attention (one wave per (b,h,16-query tile)) -------
// Computes S^T = K*Q^T so the probability tile lands directly in the
// A-operand layout for the P*V WMMA (processing key tiles in pairs of 16).
__global__ void attn_kernel(const _Float16* __restrict__ Q,
                            const _Float16* __restrict__ K,
                            const _Float16* __restrict__ VT,
                            const int* __restrict__ amask,
                            _Float16* __restrict__ Y) {
  const int wave = blockIdx.x * (blockDim.x >> 5) + (threadIdx.x >> 5);
  const int lane = threadIdx.x & 31;
  const int ls = lane & 15, hi = lane >> 4;

  const int qt = wave & 63;
  const int h  = (wave >> 6) & 15;
  const int b  = wave >> 10;
  const size_t bh = (size_t)(b * H_ + h);
  const float scale = 0.08838834764831845f;  // 1/sqrt(128)
  const int qidx = qt * 16 + ls;

  // Q as B-operand of S^T: lane holds column q = ls, e in table order.
  const _Float16* qp = Q + (bh * T_ + qt * 16 + ls) * E_ + hi * 8;
  v16h qf[4];
#pragma unroll
  for (int ec = 0; ec < 4; ++ec) qf[ec] = combine8(*(const v8h*)(qp + ec * 32), *(const v8h*)(qp + ec * 32 + 16));

  v8f oacc[8] = {};           // 8 e-tiles of 16, C-layout (rows=q, cols=e)
  float m_run = NEG_INF, l_run = 0.0f;

  for (int kt2 = 0; kt2 <= qt; kt2 += 2) {
    v8f s0, s1;
    // ---- S^T tile for key tile kt2 ----
    {
      v8f s = {};
      const _Float16* kr = K + (bh * T_ + kt2 * 16 + ls) * E_;
#pragma unroll
      for (int ec = 0; ec < 4; ++ec) {
        v8h alo = *(const v8h*)(kr + ec * 32 + hi * 8);
        v8h ahi = *(const v8h*)(kr + ec * 32 + 16 + hi * 8);
        s = wmma_f16(combine8(alo, ahi), qf[ec], s);
      }
      const int* mb = amask + b * T_ + kt2 * 16 + hi * 8;
      v4i m0v = *(const v4i*)(mb);
      v4i m1v = *(const v4i*)(mb + 4);
#pragma unroll
      for (int j = 0; j < 8; ++j) {
        int kk = kt2 * 16 + j + 8 * hi;
        int keep = (j < 4) ? m0v[j] : m1v[j - 4];
        float v = s[j] * scale;
        if (kk > qidx || keep == 0) v = NEG_INF;
        s[j] = v;
      }
      s0 = s;
    }
    // ---- S^T tile for key tile kt2+1 (or fully masked) ----
    if (kt2 + 1 <= qt) {
      v8f s = {};
      const _Float16* kr = K + (bh * T_ + (kt2 + 1) * 16 + ls) * E_;
#pragma unroll
      for (int ec = 0; ec < 4; ++ec) {
        v8h alo = *(const v8h*)(kr + ec * 32 + hi * 8);
        v8h ahi = *(const v8h*)(kr + ec * 32 + 16 + hi * 8);
        s = wmma_f16(combine8(alo, ahi), qf[ec], s);
      }
      const int* mb = amask + b * T_ + (kt2 + 1) * 16 + hi * 8;
      v4i m0v = *(const v4i*)(mb);
      v4i m1v = *(const v4i*)(mb + 4);
#pragma unroll
      for (int j = 0; j < 8; ++j) {
        int kk = (kt2 + 1) * 16 + j + 8 * hi;
        int keep = (j < 4) ? m0v[j] : m1v[j - 4];
        float v = s[j] * scale;
        if (kk > qidx || keep == 0) v = NEG_INF;
        s[j] = v;
      }
      s1 = s;
    } else {
#pragma unroll
      for (int j = 0; j < 8; ++j) s1[j] = NEG_INF;
    }

    // ---- online softmax (per q = ls, duplicated across lane halves) ----
    float mloc = NEG_INF;
#pragma unroll
    for (int j = 0; j < 8; ++j) {
      mloc = fmaxf(mloc, s0[j]);
      mloc = fmaxf(mloc, s1[j]);
    }
    float mpair = fmaxf(mloc, __shfl_xor(mloc, 16, 32));
    float m_new = fmaxf(m_run, mpair);
    float alpha = __expf(m_run - m_new);

    v8h p0h, p1h;
    float lloc = 0.0f;
#pragma unroll
    for (int j = 0; j < 8; ++j) {
      float p0 = __expf(s0[j] - m_new);
      float p1 = __expf(s1[j] - m_new);
      lloc += p0 + p1;
      p0h[j] = (_Float16)p0;
      p1h[j] = (_Float16)p1;
    }
    float lpair = lloc + __shfl_xor(lloc, 16, 32);
    l_run = l_run * alpha + lpair;
    m_run = m_new;

    // rescale O rows: row of VGPR j for this lane is q = j + 8*hi
    float ar[8];
#pragma unroll
    for (int j = 0; j < 8; ++j) ar[j] = __shfl(alpha, j + 8 * hi, 32);
#pragma unroll
    for (int et = 0; et < 8; ++et) {
#pragma unroll
      for (int j = 0; j < 8; ++j) oacc[et][j] *= ar[j];
    }

    // ---- P * V over 32 keys (tiles kt2, kt2+1) ----
    v16h af = combine8(p0h, p1h);  // A = P (16q x 32kk), exact A-layout
    const _Float16* vb = VT + (bh * E_ + ls) * T_ + kt2 * 16 + hi * 8;
#pragma unroll
    for (int et = 0; et < 8; ++et) {
      v16h bf = combine8(*(const v8h*)(vb + (size_t)et * 16 * T_), *(const v8h*)(vb + (size_t)et * 16 * T_ + 16));
      oacc[et] = wmma_f16(af, bf, oacc[et]);
    }
  }

  // ---- normalize and store to Y[b][t][h*E + e] (f16) ----
  float lr[8];
#pragma unroll
  for (int j = 0; j < 8; ++j) lr[j] = __shfl(l_run, j + 8 * hi, 32);
  _Float16* yp = Y + ((size_t)b * T_ + qt * 16 + hi * 8) * D_ + h * E_ + ls;
#pragma unroll
  for (int et = 0; et < 8; ++et) {
#pragma unroll
    for (int j = 0; j < 8; ++j) {
      yp[(size_t)j * D_ + et * 16] = (_Float16)(oacc[et][j] / lr[j]);
    }
  }
}

// ---------------- output projection: OUT = Y * Wo^T + bo (f32 out) --------
// Same 128M x 64N block tiling with async-staged Wo panel.
__global__ __launch_bounds__(256)
void proj_kernel(const _Float16* __restrict__ Yh,
                 const _Float16* __restrict__ WO,
                 const float* __restrict__ bo,
                 float* __restrict__ OUT) {
  __shared__ alignas(64) _Float16 bt[2][64 * 32];

  const int tid = threadIdx.x;
  const int wv = tid >> 5, lane = tid & 31;
  const int ls = lane & 15, hi = lane >> 4;

  const int nb = blockIdx.x & 31;        // D/64 = 32
  const int mb = blockIdx.x >> 5;        // (B*T)/128 = 32

  const int n0 = nb * 64;
  const int m0 = mb * 128 + wv * 16;

  const int crow = tid >> 2, cseg = tid & 3;
  const _Float16* gsrc = WO + ((size_t)(n0 + crow)) * D_ + cseg * 8;
  _Float16* l0 = &bt[0][crow * 32 + cseg * 8];
  _Float16* l1 = &bt[1][crow * 32 + cseg * 8];

  const _Float16* arow = Yh + ((size_t)m0 + ls) * D_;

  v8f acc[4] = {};
  constexpr int STEPS = D_ / 32;  // 64

  async_copy_b128(gsrc, l0);
  for (int s = 0; s < STEPS; ++s) {
    const int cur = s & 1;
    const bool more = (s + 1) < STEPS;
    if (more) async_copy_b128(gsrc + (size_t)(s + 1) * 32, cur ? l0 : l1);
    if (more) wait_async_le1(); else wait_async_le0();
    __syncthreads();

    const int k0 = s * 32;
    v8h alo = *(const v8h*)(arow + k0 + hi * 16);
    v8h ahi = *(const v8h*)(arow + k0 + hi * 16 + 8);
    v16h a  = combine8(alo, ahi);
#pragma unroll
    for (int nt = 0; nt < 4; ++nt) {
      v16h bb = *(const v16h*)&bt[cur][(nt * 16 + ls) * 32 + hi * 16];
      acc[nt] = wmma_f16(a, bb, acc[nt]);
    }
    __syncthreads();
  }

#pragma unroll
  for (int nt = 0; nt < 4; ++nt) {
    const int n = n0 + nt * 16 + ls;
    const float bval = bo[n];
    float* op = OUT + ((size_t)m0 + hi * 8) * D_ + n;
#pragma unroll
    for (int j = 0; j < 8; ++j) op[(size_t)j * D_] = acc[nt][j] + bval;
  }
}

// ---------------- host-side launch ----------------
extern "C" void kernel_launch(void* const* d_in, const int* in_sizes, int n_in,
                              void* d_out, int out_size, void* d_ws, size_t ws_size,
                              hipStream_t stream) {
  const float* x  = (const float*)d_in[0];
  const int*  am  = (const int*)  d_in[1];
  const float* Wq = (const float*)d_in[2];
  const float* bq = (const float*)d_in[3];
  const float* Wk = (const float*)d_in[4];
  const float* bk = (const float*)d_in[5];
  const float* Wv = (const float*)d_in[6];
  const float* bv = (const float*)d_in[7];
  const float* Wo = (const float*)d_in[8];
  const float* bo = (const float*)d_in[9];
  float* out = (float*)d_out;

  // workspace layout (f16 element offsets)
  constexpr size_t NX  = (size_t)B_ * T_ * D_;        // 8388608
  constexpr size_t NW  = (size_t)H_ * E_ * D_;        // 4194304
  constexpr size_t NWO = (size_t)D_ * D_;             // 4194304
  constexpr size_t NQ  = (size_t)B_ * H_ * T_ * E_;   // 8388608

  _Float16* wsp = (_Float16*)d_ws;
  _Float16* xh  = wsp;                 // NX
  _Float16* wqh = xh  + NX;            // NW
  _Float16* wkh = wqh + NW;            // NW
  _Float16* wvh = wkh + NW;            // NW
  _Float16* woh = wvh + NW;            // NWO
  _Float16* qh  = woh + NWO;           // NQ
  _Float16* kh  = qh  + NQ;            // NQ
  _Float16* vth = kh  + NQ;            // NQ  ([b][h][e][t])
  _Float16* yh  = vth + NQ;            // NX

  // 1) convert f32 -> f16
  cvt_f16_kernel<<<2048, 256, 0, stream>>>(x,  xh,  (int)NX);
  cvt_f16_kernel<<<1024, 256, 0, stream>>>(Wq, wqh, (int)NW);
  cvt_f16_kernel<<<1024, 256, 0, stream>>>(Wk, wkh, (int)NW);
  cvt_f16_kernel<<<1024, 256, 0, stream>>>(Wv, wvh, (int)NW);
  cvt_f16_kernel<<<1024, 256, 0, stream>>>(Wo, woh, (int)NWO);

  // 2) QKV projections: B*H*(T/128)*(E/64) = 1024 blocks of 8 waves
  qkv_gemm_kernel<<<1024, 256, 0, stream>>>(xh, wqh, bq, qh,  0);
  qkv_gemm_kernel<<<1024, 256, 0, stream>>>(xh, wkh, bk, kh,  0);
  qkv_gemm_kernel<<<1024, 256, 0, stream>>>(xh, wvh, bv, vth, 1);

  // 3) flash attention: 4096 waves -> 512 blocks of 256
  attn_kernel<<<512, 256, 0, stream>>>(qh, kh, vth, am, yh);

  // 4) output projection: (B*T/128)*(D/64) = 1024 blocks of 8 waves
  proj_kernel<<<1024, 256, 0, stream>>>(yh, woh, bo, out);
}
